// NTXentLoss_3229815406857
// MI455X (gfx1250) — matrix operand order repaired and checked by reading, not compiled, not run on hardware
//
#include <hip/hip_runtime.h>
#include <cstdint>
#include <cstddef>

// ---------- problem constants ----------
constexpr int   N_    = 4096;
constexpr int   D_    = 512;
constexpr int   TWO_N = 8192;
constexpr float INV_T = 10.0f;   // 1 / temperature
constexpr float EPS_  = 1e-8f;

// ---------- workspace layout (bytes) ----------
constexpr size_t OFF_INV  = 0;                        // 8192 f32 inv-norms
constexpr size_t OFF_POS  = 32768;                    // 8192 f32 positive logits
constexpr size_t OFF_PART = 65536;                    // 512 f32 per-block partial losses
constexpr size_t OFF_RHI  = 131072;                   // 8192x512 bf16 (hi)
constexpr size_t OFF_RLO  = OFF_RHI + (size_t)TWO_N * D_ * 2;  // 8192x512 bf16 (lo)

// ---------- types ----------
typedef __attribute__((ext_vector_type(16))) __bf16 bf16x16;
typedef __attribute__((ext_vector_type(8)))  float  v8f;

union FragU { uint32_t u[8]; bf16x16 v; };

// ---------- helpers ----------
__device__ __forceinline__ unsigned short f2bf(float f) {
  union { float f; uint32_t u; } c; c.f = f;
  uint32_t u = c.u;
  uint32_t r = u + 0x7FFFu + ((u >> 16) & 1u);   // round-to-nearest-even
  return (unsigned short)(r >> 16);
}
__device__ __forceinline__ float bf2f(unsigned short h) {
  union { uint32_t u; float f; } c; c.u = ((uint32_t)h) << 16;
  return c.f;
}
__device__ __forceinline__ bf16x16 load_frag(const unsigned short* p0,
                                             const unsigned short* p1) {
  FragU f;
  uint4 s0 = *reinterpret_cast<const uint4*>(p0);   // 16B contiguous (8 bf16)
  uint4 s1 = *reinterpret_cast<const uint4*>(p1);   // 16B contiguous (8 bf16)
  f.u[0] = s0.x; f.u[1] = s0.y; f.u[2] = s0.z; f.u[3] = s0.w;
  f.u[4] = s1.x; f.u[5] = s1.y; f.u[6] = s1.z; f.u[7] = s1.w;
  return f.v;
}
// online logsumexp merge of (m,s) with one sample x
__device__ __forceinline__ void lse_push(float& m, float& s, float x) {
  float nm = fmaxf(m, x);
  s = s * __expf(m - nm) + __expf(x - nm);
  m = nm;
}
// merge two (m,s) states
__device__ __forceinline__ void lse_merge(float& m, float& s, float om, float os) {
  float nm = fmaxf(m, om);
  s = s * __expf(m - nm) + os * __expf(om - nm);
  m = nm;
}

// ---------- kernel 1: normalize rows, split into bf16 hi/lo ----------
__global__ void __launch_bounds__(256)
ntx_prep(const float* __restrict__ zi, const float* __restrict__ zj,
         unsigned short* __restrict__ rhi, unsigned short* __restrict__ rlo,
         float* __restrict__ invnorm) {
  const int r = blockIdx.x;                       // 0..8191
  const float* src = (r < N_) ? (zi + (size_t)r * D_)
                              : (zj + (size_t)(r - N_) * D_);
  const int t = threadIdx.x;                      // 0..255, handles elems t, t+256
  float v0 = src[t];
  float v1 = src[t + 256];
  float loc = v0 * v0 + v1 * v1;
  for (int m = 16; m >= 1; m >>= 1) loc += __shfl_xor(loc, m, 32);
  __shared__ float wsum[8];
  __shared__ float s_inv;
  if ((t & 31) == 0) wsum[t >> 5] = loc;
  __syncthreads();
  if (t == 0) {
    float s = 0.f;
    for (int i = 0; i < 8; ++i) s += wsum[i];
    float inv = 1.0f / fmaxf(sqrtf(s), EPS_);     // torch eps clamp
    s_inv = inv;
    invnorm[r] = inv;
  }
  __syncthreads();
  const float inv = s_inv;
  const size_t base = (size_t)r * D_;
  float x0 = v0 * inv;
  float x1 = v1 * inv;
  unsigned short h0 = f2bf(x0);
  unsigned short h1 = f2bf(x1);
  rhi[base + t]       = h0;
  rhi[base + t + 256] = h1;
  rlo[base + t]       = f2bf(x0 - bf2f(h0));
  rlo[base + t + 256] = f2bf(x1 - bf2f(h1));
}

// ---------- kernel 2: positive-pair logits (fp32 exact dot) ----------
__global__ void __launch_bounds__(256)
ntx_pos(const float* __restrict__ zi, const float* __restrict__ zj,
        const float* __restrict__ invnorm, float* __restrict__ pos) {
  const int r = blockIdx.x;                       // 0..4095
  const int t = threadIdx.x;
  const float* a = zi + (size_t)r * D_;
  const float* b = zj + (size_t)r * D_;
  float loc = a[t] * b[t] + a[t + 256] * b[t + 256];
  for (int m = 16; m >= 1; m >>= 1) loc += __shfl_xor(loc, m, 32);
  __shared__ float wsum[8];
  if ((t & 31) == 0) wsum[t >> 5] = loc;
  __syncthreads();
  if (t == 0) {
    float s = 0.f;
    for (int i = 0; i < 8; ++i) s += wsum[i];
    float p = s * invnorm[r] * invnorm[r + N_] * INV_T;
    pos[r]      = p;                               // sim_ij == sim_ji (symmetric)
    pos[r + N_] = p;
  }
}

// ---------- kernel 3: fused sim-GEMM + online logsumexp ----------
// 512 blocks, each owns 16 rows; 8 waves, each one 16x16 WMMA tile; BN=128.
__global__ void __launch_bounds__(256, 1)
ntx_main(const unsigned short* __restrict__ rhi,
         const unsigned short* __restrict__ rlo,
         const float* __restrict__ pos,
         float* __restrict__ partials) {
  const int lane = threadIdx.x & 31;
  const int wave = threadIdx.x >> 5;              // 0..7
  const int half = (lane < 16) ? 0 : 1;
  const int rb   = blockIdx.x * 16;               // row-tile base

  // A fragment: lane holds row rb+(lane&15); two 8-elem segments per k-chunk.
  const int mrow = rb + (lane & 15);
  const int a_k0 = half ? 16 : 0;
  const int a_k1 = half ? 24 : 8;
  // B fragment: lane holds Rn row (column) cwb+(lane&15); one 16-elem run.
  const int b_k0 = half ? 16 : 0;

  const unsigned short* arow_h = rhi + (size_t)mrow * D_;
  const unsigned short* arow_l = rlo + (size_t)mrow * D_;

  float mx[8], sm[8];
#pragma unroll
  for (int g = 0; g < 8; ++g) { mx[g] = -INFINITY; sm[g] = 0.0f; }

  for (int cb = 0; cb < TWO_N; cb += 128) {
    const int cwb  = cb + wave * 16;
    const int gcol = cwb + (lane & 15);
    const unsigned short* brow_h = rhi + (size_t)gcol * D_;
    const unsigned short* brow_l = rlo + (size_t)gcol * D_;

    v8f acc = {0.f, 0.f, 0.f, 0.f, 0.f, 0.f, 0.f, 0.f};
#pragma unroll 4
    for (int kc = 0; kc < D_; kc += 32) {
      bf16x16 ah = load_frag(arow_h + kc + a_k0, arow_h + kc + a_k1);
      bf16x16 al = load_frag(arow_l + kc + a_k0, arow_l + kc + a_k1);
      bf16x16 bh = load_frag(brow_h + kc + b_k0, brow_h + kc + b_k0 + 8);
      bf16x16 bl = load_frag(brow_l + kc + b_k0, brow_l + kc + b_k0 + 8);
      // bf16x3 compensated product: hi*hi + hi*lo + lo*hi  (lo*lo negligible)
      acc = __builtin_amdgcn_wmma_f32_16x16x32_bf16(false, ah, false, bh,
                                                    (short)0, acc, false, false);
      acc = __builtin_amdgcn_wmma_f32_16x16x32_bf16(false, ah, false, bl,
                                                    (short)0, acc, false, false);
      acc = __builtin_amdgcn_wmma_f32_16x16x32_bf16(false, al, false, bh,
                                                    (short)0, acc, false, false);
    }
    // C/D layout: VGPR g -> rows rb+g (lanes 0-15) / rb+8+g (lanes 16-31),
    // column = cwb + (lane&15).
#pragma unroll
    for (int g = 0; g < 8; ++g) {
      const int grow = rb + g + (half ? 8 : 0);
      float x = acc[g] * INV_T;
      if (grow == gcol) x = 0.0f;                 // zeroed self-similarity logit
      lse_push(mx[g], sm[g], x);
    }
  }

  // ---- reduce across the 16 lanes of each half (same rows per half) ----
#pragma unroll
  for (int mask = 8; mask >= 1; mask >>= 1) {
#pragma unroll
    for (int g = 0; g < 8; ++g) {
      float om = __shfl_xor(mx[g], mask, 32);
      float os = __shfl_xor(sm[g], mask, 32);
      lse_merge(mx[g], sm[g], om, os);
    }
  }

  __shared__ float red_m[8][16];
  __shared__ float red_s[8][16];
  __shared__ float losses[16];
  if (lane == 0) {
#pragma unroll
    for (int g = 0; g < 8; ++g) { red_m[wave][g] = mx[g]; red_s[wave][g] = sm[g]; }
  }
  if (lane == 16) {
#pragma unroll
    for (int g = 0; g < 8; ++g) { red_m[wave][8 + g] = mx[g]; red_s[wave][8 + g] = sm[g]; }
  }
  __syncthreads();

  if (threadIdx.x < 16) {
    const int t = threadIdx.x;
    float m = red_m[0][t], s = red_s[0][t];
    for (int w = 1; w < 8; ++w) lse_merge(m, s, red_m[w][t], red_s[w][t]);
    const float p = pos[rb + t];                  // positive logit (column 0)
    lse_push(m, s, p);                            // 2N+1 logits total
    losses[t] = (m + __logf(s)) - p;              // -log_softmax[0]
  }
  __syncthreads();
  if (threadIdx.x == 0) {
    float a = 0.f;
    for (int i = 0; i < 16; ++i) a += losses[i];  // fixed order: deterministic
    partials[blockIdx.x] = a;
  }
}

// ---------- kernel 4: deterministic final reduction ----------
__global__ void ntx_finalize(const float* __restrict__ partials,
                             float* __restrict__ out) {
  float s = 0.f;
  for (int i = 0; i < TWO_N / 16; ++i) s += partials[i];
  out[0] = s / (float)TWO_N;
}

// ---------- launcher ----------
extern "C" void kernel_launch(void* const* d_in, const int* in_sizes, int n_in,
                              void* d_out, int out_size, void* d_ws, size_t ws_size,
                              hipStream_t stream) {
  (void)in_sizes; (void)n_in; (void)out_size; (void)ws_size;
  const float* zi = (const float*)d_in[0];
  const float* zj = (const float*)d_in[1];
  float* out = (float*)d_out;

  char* ws = (char*)d_ws;
  float*          invnorm  = (float*)(ws + OFF_INV);
  float*          pos      = (float*)(ws + OFF_POS);
  float*          partials = (float*)(ws + OFF_PART);
  unsigned short* rhi      = (unsigned short*)(ws + OFF_RHI);
  unsigned short* rlo      = (unsigned short*)(ws + OFF_RLO);

  ntx_prep<<<TWO_N, 256, 0, stream>>>(zi, zj, rhi, rlo, invnorm);
  ntx_pos<<<N_, 256, 0, stream>>>(zi, zj, invnorm, pos);
  ntx_main<<<TWO_N / 16, 256, 0, stream>>>(rhi, rlo, pos, partials);
  ntx_finalize<<<1, 1, 0, stream>>>(partials, out);
}
